// AttentionHead_42502996361619
// MI455X (gfx1250) — matrix operand order repaired and checked by reading, not compiled, not run on hardware
//
#include <hip/hip_runtime.h>
#include <stdint.h>

#define BB  4
#define SS  4096
#define DM  256
#define DQK 64

typedef __attribute__((ext_vector_type(16))) _Float16 v16h;
typedef __attribute__((ext_vector_type(8)))  float    v8f;

// --- CDNA5 async copy: 16 bytes per lane, global -> LDS, tracked by ASYNCcnt.
__device__ __forceinline__ void async_cp16(uint32_t lds_byte, uint64_t gaddr) {
  asm volatile("global_load_async_to_lds_b128 %0, %1, off"
               :: "v"(lds_byte), "v"(gaddr) : "memory");
}
__device__ __forceinline__ void wait_async0() {
#if __has_builtin(__builtin_amdgcn_s_wait_asynccnt)
  __builtin_amdgcn_s_wait_asynccnt(0);
  asm volatile("" ::: "memory");
#else
  asm volatile("s_wait_asynccnt 0" ::: "memory");
#endif
}

// ---------------------------------------------------------------------------
// Projection: Y = X[f32,MxK] @ W^T (W [NxK] fp32), f16 output.
// TRANSPOSE=true writes Y as [B][N][S] (used for V so attention can stage
// V^T rows with contiguous async copies).
// ---------------------------------------------------------------------------
template <bool TRANSPOSE>
__global__ __launch_bounds__(256) void proj_f16_wmma(
    const float* __restrict__ X, const float* __restrict__ W,
    _Float16* __restrict__ Y, int M, int N, int K) {
  const int wave = (int)((blockIdx.x * blockDim.x + threadIdx.x) >> 5);
  const int lane = (int)(threadIdx.x & 31);
  const int ntiles = N >> 4;
  const int mt = wave / ntiles;
  const int nt = wave - mt * ntiles;
  if (mt >= (M >> 4)) return;
  const int r16 = lane & 15;
  const int hi  = lane >> 4;

  v8f acc = {};
  const float* arow = X + (size_t)(mt * 16 + r16) * K;
  const float* brow = W + (size_t)(nt * 16 + r16) * K;
  for (int kb = 0; kb < K; kb += 32) {
    v16h a, bm;
    const float* ap = arow + kb + hi * 16;
#pragma unroll
    for (int i = 0; i < 8; ++i) {
      a[i]     = (_Float16)ap[i];
      a[i + 8] = (_Float16)ap[8 + i];
    }
    const float* bp = brow + kb + hi * 16;
#pragma unroll
    for (int i = 0; i < 16; ++i) bm[i] = (_Float16)bp[i];
    acc = __builtin_amdgcn_wmma_f32_16x16x32_f16(false, a, false, bm,
                                                 (short)0, acc, false, false);
  }
  if (TRANSPOSE) {
    const int row0 = mt * 16 + hi * 8;
    const int bb   = row0 >> 12;              // tile never straddles batches
    const int col  = nt * 16 + r16;
    _Float16* yp = Y + ((size_t)bb * DM + col) * SS + (row0 & (SS - 1));
#pragma unroll
    for (int r = 0; r < 8; ++r) yp[r] = (_Float16)acc[r];
  } else {
    _Float16* yp = Y + (size_t)(mt * 16 + hi * 8) * N + nt * 16 + r16;
#pragma unroll
    for (int r = 0; r < 8; ++r) yp[(size_t)r * N] = (_Float16)acc[r];
  }
}

// ---------------------------------------------------------------------------
// Flash attention, 64-key chunks, double-buffered async global->LDS staging.
// Block = 8 waves = 128 query rows; wave holds 16x64 Q as A-frags and
// O (16x256) as 16 f32 C-tiles. All causal guards are SGPR compares
// (readfirstlane) so WMMA always runs with EXEC == all-1s.
// Dynamic LDS layout (halves):
//   [0, 8192)       K double buffer   (2 x 64x64)
//   [8192, 40960)   V^T double buffer (2 x 256x64)
//   [40960, 49152)  per-wave P scratch (8 x 16x64)
// ---------------------------------------------------------------------------
__global__ __launch_bounds__(256) void flash_attn_wmma(
    const _Float16* __restrict__ Qh, const _Float16* __restrict__ Kh,
    const _Float16* __restrict__ VTh, float* __restrict__ Out) {
  extern __shared__ _Float16 smem[];
  const int tid  = (int)threadIdx.x;
  const int lane = tid & 31;
  const int w    = tid >> 5;
  const int r16  = lane & 15;
  const int hi   = lane >> 4;

  const int b      = (int)blockIdx.x >> 5;
  const int qblock = ((int)blockIdx.x & 31) * 128;
  // Wave-uniform query base as an SGPR -> scalar branches for causal skips.
  const int qg = __builtin_amdgcn_readfirstlane(qblock + w * 16);
  const uint32_t ldsb = (uint32_t)(uintptr_t)smem;

  // Q tile (16x64) as two 16x32 A fragments.
  const _Float16* qp = Qh + ((size_t)b * SS + qg + r16) * DQK;
  v16h qf[2];
#pragma unroll
  for (int c = 0; c < 2; ++c) {
    const int base = c * 32 + hi * 16;
#pragma unroll
    for (int i = 0; i < 8; ++i) {
      qf[c][i]     = qp[base + i];
      qf[c][i + 8] = qp[base + 8 + i];
    }
  }

  v8f o[16] = {};
  float m[8], l[8];
#pragma unroll
  for (int r = 0; r < 8; ++r) { m[r] = -3.0e38f; l[r] = 0.f; }

  auto stage = [&](int buf, int kb) {
    const uint64_t kg = (uint64_t)(uintptr_t)(Kh + ((size_t)b * SS + kb) * DQK);
    const uint32_t kl = ldsb + (uint32_t)buf * (4096u * 2u);
#pragma unroll
    for (int u = 0; u < 2; ++u) {                 // K: 8KB, contiguous
      const uint32_t f = (uint32_t)(u * 256 + tid) * 16u;
      async_cp16(kl + f, kg + f);
    }
    const uint32_t vl = ldsb + 8192u * 2u + (uint32_t)buf * (16384u * 2u);
#pragma unroll
    for (int u = 0; u < 8; ++u) {                 // V^T: 32KB, 128B rows
      const uint32_t f   = (uint32_t)(u * 256 + tid) * 16u;
      const uint32_t n   = f >> 7;                // 128 bytes per n-row
      const uint32_t off = f & 127u;
      const uint64_t vg =
          (uint64_t)(uintptr_t)(VTh + ((size_t)b * DM + n) * SS + kb) + off;
      async_cp16(vl + f, vg);
    }
  };

  const int limit = qblock + 128;                 // block causal frontier
  stage(0, 0);
  for (int kbase = 0, it = 0; kbase < limit; kbase += 64, ++it) {
    const int buf = it & 1;
    wait_async0();                                // staged data landed
    __syncthreads();                              // visible to all waves
    if (kbase + 64 < limit) stage(buf ^ 1, kbase + 64);  // overlap next DMA

    if (kbase <= qg + 15) {                       // scalar (SGPR) causal skip
      const _Float16* ksp = smem + (size_t)buf * 4096;
      const _Float16* vtp = smem + 8192 + (size_t)buf * 16384;
      _Float16*       pw  = smem + 40960 + (size_t)w * 1024;

      // S (16x64) = Q(16x64) @ K^T : four 16x16 C-tiles, scalar per-tile skip.
      v8f st[4] = {};
#pragma unroll
      for (int jt = 0; jt < 4; ++jt) {
        if (kbase + jt * 16 <= qg + 15) {
#pragma unroll
          for (int c = 0; c < 2; ++c) {
            v16h bk;
            const _Float16* kp =
                ksp + (size_t)(jt * 16 + r16) * DQK + c * 32 + hi * 16;
#pragma unroll
            for (int i = 0; i < 16; ++i) bk[i] = kp[i];
            st[jt] = __builtin_amdgcn_wmma_f32_16x16x32_f16(
                false, qf[c], false, bk, (short)0, st[jt], false, false);
          }
        }
      }

      // Online softmax over 64 fresh columns; P written straight to LDS.
#pragma unroll
      for (int r = 0; r < 8; ++r) {
        const int qrow = qg + hi * 8 + r;
        float x[4];
#pragma unroll
        for (int jt = 0; jt < 4; ++jt) {
          const int j = kbase + jt * 16 + r16;
          x[jt] = (j <= qrow) ? st[jt][r] * 0.125f : -1.0e9f;
        }
        float tmax = fmaxf(fmaxf(x[0], x[1]), fmaxf(x[2], x[3]));
        tmax = fmaxf(tmax, __shfl_xor(tmax, 1));
        tmax = fmaxf(tmax, __shfl_xor(tmax, 2));
        tmax = fmaxf(tmax, __shfl_xor(tmax, 4));
        tmax = fmaxf(tmax, __shfl_xor(tmax, 8));
        const float mnew  = fmaxf(m[r], tmax);
        const float alpha = __expf(m[r] - mnew);
        float sr = 0.f;
#pragma unroll
        for (int jt = 0; jt < 4; ++jt) {
          const float e = __expf(x[jt] - mnew);
          sr += e;
          pw[(hi * 8 + r) * 64 + jt * 16 + r16] = (_Float16)e;
        }
        sr += __shfl_xor(sr, 1);
        sr += __shfl_xor(sr, 2);
        sr += __shfl_xor(sr, 4);
        sr += __shfl_xor(sr, 8);
        l[r] = l[r] * alpha + sr;
        m[r] = mnew;
#pragma unroll
        for (int t = 0; t < 16; ++t) o[t][r] *= alpha;
      }

      // O(16x256) += P(16x64) @ V(64x256); scalar skip of masked upper half.
#pragma unroll
      for (int c = 0; c < 2; ++c) {
        if (c == 0 || kbase + 32 <= qg + 15) {
          v16h pa;   // DS ops are in-order within a wave
          const _Float16* pr = pw + (size_t)r16 * 64 + c * 32 + hi * 16;
#pragma unroll
          for (int i = 0; i < 8; ++i) { pa[i] = pr[i]; pa[i + 8] = pr[8 + i]; }
#pragma unroll
          for (int t = 0; t < 16; ++t) {
            v16h bv;
            const _Float16* vp =
                vtp + (size_t)(t * 16 + r16) * 64 + c * 32 + hi * 16;
#pragma unroll
            for (int i = 0; i < 16; ++i) bv[i] = vp[i];
            o[t] = __builtin_amdgcn_wmma_f32_16x16x32_f16(
                false, pa, false, bv, (short)0, o[t], false, false);
          }
        }
      }
    }
  }

  // Normalize and store fp32 output.
#pragma unroll
  for (int r = 0; r < 8; ++r) {
    const float inv = (l[r] > 0.f) ? 1.f / l[r] : 0.f;
    float* op = Out + ((size_t)b * SS + qg + hi * 8 + r) * DM + r16;
#pragma unroll
    for (int t = 0; t < 16; ++t) op[t * 16] = o[t][r] * inv;
  }
}

// ---------------------------------------------------------------------------
extern "C" void kernel_launch(void* const* d_in, const int* in_sizes, int n_in,
                              void* d_out, int out_size, void* d_ws, size_t ws_size,
                              hipStream_t stream) {
  const float* enc_q = (const float*)d_in[0];
  const float* enc_k = (const float*)d_in[1];
  const float* enc_v = (const float*)d_in[2];
  // d_in[3] = boolean causal mask: recomputed analytically in-kernel.
  const float* Wq = (const float*)d_in[4];
  const float* Wk = (const float*)d_in[5];
  const float* Wv = (const float*)d_in[6];
  float* out = (float*)d_out;

  _Float16* qh = (_Float16*)d_ws;                       // [B*S][64]
  _Float16* kh = qh + (size_t)BB * SS * DQK;            // [B*S][64]
  _Float16* vh = kh + (size_t)BB * SS * DQK;            // V^T: [B][256][S]

  const int Mrows = BB * SS;
  {
    const int waves = (Mrows / 16) * (DQK / 16);        // 4096
    proj_f16_wmma<false><<<waves / 8, 256, 0, stream>>>(enc_q, Wq, qh, Mrows, DQK, DM);
    proj_f16_wmma<false><<<waves / 8, 256, 0, stream>>>(enc_k, Wk, kh, Mrows, DQK, DM);
  }
  {
    const int waves = (Mrows / 16) * (DM / 16);         // 16384
    proj_f16_wmma<true><<<waves / 8, 256, 0, stream>>>(enc_v, Wv, vh, Mrows, DM, DM);
  }
  flash_attn_wmma<<<BB * (SS / 128), 256, 96 * 1024, stream>>>(qh, kh, vh, out);
}
